// LearnedSimulator_18236431138857
// MI455X (gfx1250) — matrix operand order repaired and checked by reading, not compiled, not run on hardware
//
#include <hip/hip_runtime.h>

// ============================================================================
// CircuitGNN forward for MI455X (gfx1250), wave32 + WMMA f16 16x16x32.
//
// Pipeline (all on `stream`):
//   1. k_node_encode : node = LN-MLP([embed[type], feat] pad 32) ; zero aggr
//   2. k_edge_in     : ef   = LN-MLP(edge_attr pad 32)           (f16, 51MB ws)
//   3. 4x { k_mp_edge : m = LN-MLP([x_i,x_j,ef] K=96); ef+=m; fp-atomic aggr
//           k_mp_node : x += LN-MLP([x,aggr] K=64); zero aggr }
//   4. k_node_out    : out = MLP(x) (32->32->32->1)              ; zero netI
//   5. k_circuit     : per-edge diode current, fp-atomic netI[dst]
//   6. k_node_err    : node_error from netI & y
//
// GEMMs: v_wmma_f32_16x16x32_f16, one wave per 16-row tile. Hidden activations
// are restaged through LDS in a K-permuted order pi(c)=2*(c&15)+(c>>4) so the
// C-fragment's two per-lane outputs pack into a single ds_store_b32; the next
// layer's weights are staged with the same K permutation (WMMA is invariant
// under a shared K-permutation of A and B). Segment-sums use unsafeAtomicAdd
// (native global_atomic_add_f32). Workspace required: ~68 MB.
// ============================================================================

typedef _Float16 h16;
typedef __attribute__((ext_vector_type(16))) _Float16 v16h;
typedef __attribute__((ext_vector_type(4)))  _Float16 h16x4;
typedef __attribute__((ext_vector_type(8)))  float    v8f;
typedef __attribute__((ext_vector_type(4)))  float    f32x4;
typedef __attribute__((ext_vector_type(4)))  unsigned int u32x4;

#define N_NODES 50000
#define N_EDGES 800000
#define LN_EPS  1e-5f
#define VTH     0.02568f

#define NODE_TILES  3125    // 50000 / 16
#define NODE_BLOCKS 782     // ceil(3125 / 4 waves)
#define EDGE_BLOCKS 12500   // 800000 / (4 waves * 16 rows)

struct MlpW { const float *W0,*W1,*W2,*b0,*b1,*b2,*be,*g; };

// ---------------------------------------------------------------------------
// WMMA helpers
// ---------------------------------------------------------------------------
__device__ __forceinline__ v8f wmma16(v16h a, v16h b, v8f c) {
  return __builtin_amdgcn_wmma_f32_16x16x32_f16(false, a, false, b,
                                                (short)0, c, false, false);
}

// A fragment 16x32 f16 (ISA 7.12.2): lanes 0-15 rows M=0..15 hold K = kb+{0..7,
// 16..23}; lanes 16-31 hold K = kb+{8..15,24..31}. Two b128 LDS loads.
__device__ __forceinline__ v16h load_A(const h16* s, int stride, int kb, int lane) {
  const int row = lane & 15, g = lane >> 4;
  const h16* p = s + row * stride + kb + 8 * g;
  union { v16h v; u32x4 q[2]; } u;
  u.q[0] = *(const u32x4*)(p);        // K = kb + 8g + 0..7
  u.q[1] = *(const u32x4*)(p + 16);   // K = kb + 16 + 8g + 0..7
  return u.v;
}

// B fragment 32x16 f16: lane holds column n, K = kb + 8g + {0..7, 16..23}.
// Weights staged transposed in LDS as wt[n][Kpad] so this is 2 b128 loads.
__device__ __forceinline__ v16h load_B(const h16* wt, int Kpad, int kb,
                                       int ntile, int lane) {
  const int n = (lane & 15) + ntile * 16, g = lane >> 4;
  const h16* p = wt + n * Kpad + kb + 8 * g;
  union { v16h v; u32x4 q[2]; } u;
  u.q[0] = *(const u32x4*)(p);
  u.q[1] = *(const u32x4*)(p + 16);
  return u.v;
}

__device__ __forceinline__ void cp32h(h16* dst, const h16* src) {  // 64B
  const u32x4* s = (const u32x4*)src; u32x4* d = (u32x4*)dst;
  d[0] = s[0]; d[1] = s[1]; d[2] = s[2]; d[3] = s[3];
}
__device__ __forceinline__ void cp16h(h16* dst, const h16* src) {  // 32B
  const u32x4* s = (const u32x4*)src; u32x4* d = (u32x4*)dst;
  d[0] = s[0]; d[1] = s[1];
}
__device__ __forceinline__ h16x4 to_h4(f32x4 v) {
  h16x4 r; r[0] = (h16)v[0]; r[1] = (h16)v[1]; r[2] = (h16)v[2]; r[3] = (h16)v[3];
  return r;
}

// C-fragment epilogue for hidden layers: bias + relu, stored K-PERMUTED so the
// two per-lane values (cols c, c+16) land adjacent -> one packed b32 store.
// Stored index of hidden col c is pi(c) = 2*(c&15) + (c>>4).
__device__ __forceinline__ void store_hidden_relu(v8f c0, v8f c1, const float* b,
                                                  h16* sT, int lane) {
  const int col = lane & 15, g = lane >> 4;
  const float bb0 = b[col], bb1 = b[col + 16];
#pragma unroll
  for (int r = 0; r < 8; ++r) {
    int row = r + 8 * g;
    union { h16 h[2]; unsigned int u; } pk;
    pk.h[0] = (h16)fmaxf(c0[r] + bb0, 0.f);
    pk.h[1] = (h16)fmaxf(c1[r] + bb1, 0.f);
    *(unsigned int*)(sT + row * 32 + col * 2) = pk.u;
  }
}

// Stage one 3-layer MLP's weights into LDS, f32->f16, transposed [n][Kpad].
// Layers 2/3 consume the permuted hidden order: stored k -> original c(k).
__device__ void stage_mlp(const MlpW& w, int K1, int K1pad,
                          h16* w1t, h16* w2t, h16* w3t,
                          float* b1, float* b2, float* b3,
                          float* gam, float* bet, int tid, int nt) {
  for (int i = tid; i < 32 * K1pad; i += nt) {
    int n = i / K1pad, k = i - n * K1pad;
    w1t[i] = (k < K1) ? (h16)w.W0[k * 32 + n] : (h16)0.f;
  }
  for (int i = tid; i < 1024; i += nt) {
    int n = i >> 5, k = i & 31;
    int c = ((k & 1) << 4) | (k >> 1);      // inverse of pi()
    w2t[i] = (h16)w.W1[c * 32 + n];
    w3t[i] = (h16)w.W2[c * 32 + n];
  }
  for (int i = tid; i < 32; i += nt) {
    b1[i] = w.b0[i]; b2[i] = w.b1[i]; b3[i] = w.b2[i];
    gam[i] = w.g ? w.g[i] : 1.f;
    bet[i] = w.be ? w.be[i] : 0.f;
  }
}

// One 16-row tile through [K1 -> 32 -> 32 -> 32] with ReLU,ReLU,LayerNorm.
// sA: [16][K1] f16 input staging, sT: [16][32] f16 restage (permuted order),
// sF: [16][32] f32 final (original order).
template <int K1>
__device__ void mlp3_tile(const h16* sA, h16* sT, float* sF,
                          const h16* w1t, const h16* w2t, const h16* w3t,
                          const float* b1, const float* b2, const float* b3,
                          const float* gam, const float* bet, int lane) {
  const int col = lane & 15, g = lane >> 4;
  const v8f vz = {};
  v8f c0 = vz, c1 = vz;
#pragma unroll
  for (int kb = 0; kb < K1; kb += 32) {
    v16h a = load_A(sA, K1, kb, lane);
    c0 = wmma16(a, load_B(w1t, K1, kb, 0, lane), c0);
    c1 = wmma16(a, load_B(w1t, K1, kb, 1, lane), c1);
  }
  store_hidden_relu(c0, c1, b1, sT, lane);
  __syncthreads();
  {  // layer 2 (K in permuted order, w2t staged to match)
    v16h a = load_A(sT, 32, 0, lane);
    c0 = wmma16(a, load_B(w2t, 32, 0, 0, lane), vz);
    c1 = wmma16(a, load_B(w2t, 32, 0, 1, lane), vz);
  }
  __syncthreads();
  store_hidden_relu(c0, c1, b2, sT, lane);
  __syncthreads();
  {  // layer 3 (linear) -> sF f32, original column order
    v16h a = load_A(sT, 32, 0, lane);
    c0 = wmma16(a, load_B(w3t, 32, 0, 0, lane), vz);
    c1 = wmma16(a, load_B(w3t, 32, 0, 1, lane), vz);
    float bb0 = b3[col], bb1 = b3[col + 16];
#pragma unroll
    for (int r = 0; r < 8; ++r) {
      int row = r + 8 * g;
      sF[row * 32 + col]      = c0[r] + bb0;
      sF[row * 32 + col + 16] = c1[r] + bb1;
    }
  }
  __syncthreads();
  {  // layernorm over 32 cols, vectorized b128 LDS traffic
    const int row = lane & 15;
    float* fr = sF + row * 32;
    f32x4 v[8];
#pragma unroll
    for (int q = 0; q < 8; ++q) v[q] = *(const f32x4*)(fr + 4 * q);
    float mu = 0.f;
#pragma unroll
    for (int q = 0; q < 8; ++q) mu += v[q][0] + v[q][1] + v[q][2] + v[q][3];
    mu *= (1.f / 32.f);
    float var = 0.f;
#pragma unroll
    for (int q = 0; q < 8; ++q) {
#pragma unroll
      for (int j = 0; j < 4; ++j) { float d = v[q][j] - mu; var += d * d; }
    }
    var *= (1.f / 32.f);
    float inv = rsqrtf(var + LN_EPS);
    __syncthreads();
#pragma unroll
    for (int qq = 0; qq < 4; ++qq) {
      int q = 4 * g + qq;
      f32x4 t;
#pragma unroll
      for (int j = 0; j < 4; ++j) {
        int k = q * 4 + j;
        t[j] = (v[q][j] - mu) * inv * gam[k] + bet[k];
      }
      *(f32x4*)(fr + 4 * q) = t;
    }
  }
  __syncthreads();
}

// Read this lane's half-row (16 floats) of an sF row as 4 x f32x4.
__device__ __forceinline__ void read_half_row(const float* fr, int g, f32x4 out[4]) {
#pragma unroll
  for (int q = 0; q < 4; ++q) out[q] = *(const f32x4*)(fr + 16 * g + 4 * q);
}

// ---------------------------------------------------------------------------
// Kernels
// ---------------------------------------------------------------------------
__global__ __launch_bounds__(128) void k_node_encode(
    const int* __restrict__ xt, const float* __restrict__ xf,
    const float* __restrict__ embed, MlpW w,
    float* __restrict__ nf32, h16* __restrict__ nf16, float* __restrict__ aggr) {
  __shared__ h16 w1t[32 * 32], w2t[32 * 32], w3t[32 * 32];
  __shared__ float b1[32], b2[32], b3[32], gam[32], bet[32];
  __shared__ h16 sA[4][16 * 32], sT[4][16 * 32];
  __shared__ float sF[4][16 * 32];
  const int tid = threadIdx.x, wv = tid >> 5, lane = tid & 31;
  stage_mlp(w, 10, 32, w1t, w2t, w3t, b1, b2, b3, gam, bet, tid, 128);
  int tile = blockIdx.x * 4 + wv; if (tile >= NODE_TILES) tile = NODE_TILES - 1;
  const int row = lane & 15, g = lane >> 4;
  const int node = tile * 16 + row;
  h16* ar = &sA[wv][row * 32];
  if (g == 0) {
    int t = xt[node];
#pragma unroll
    for (int k = 0; k < 8; ++k) ar[k] = (h16)embed[t * 8 + k];
    ar[8] = (h16)xf[node * 2 + 0];
    ar[9] = (h16)xf[node * 2 + 1];
    for (int k = 10; k < 16; ++k) ar[k] = (h16)0.f;
  } else {
    for (int k = 16; k < 32; ++k) ar[k] = (h16)0.f;
  }
  __syncthreads();
  mlp3_tile<32>(sA[wv], sT[wv], sF[wv], w1t, w2t, w3t, b1, b2, b3, gam, bet, lane);
  const float* fr = &sF[wv][row * 32];
  f32x4 hv[4]; read_half_row(fr, g, hv);
  const f32x4 z4 = {};
#pragma unroll
  for (int q = 0; q < 4; ++q) {
    int k = 16 * g + 4 * q;
    *(f32x4*)(nf32 + (size_t)node * 32 + k) = hv[q];
    *(h16x4*)(nf16 + (size_t)node * 32 + k) = to_h4(hv[q]);
    *(f32x4*)(aggr + (size_t)node * 32 + k) = z4;
  }
}

__global__ __launch_bounds__(128) void k_edge_in(
    const float* __restrict__ ea, MlpW w, h16* __restrict__ ef) {
  __shared__ h16 w1t[32 * 32], w2t[32 * 32], w3t[32 * 32];
  __shared__ float b1[32], b2[32], b3[32], gam[32], bet[32];
  __shared__ h16 sA[4][16 * 32], sT[4][16 * 32];
  __shared__ float sF[4][16 * 32];
  const int tid = threadIdx.x, wv = tid >> 5, lane = tid & 31;
  stage_mlp(w, 5, 32, w1t, w2t, w3t, b1, b2, b3, gam, bet, tid, 128);
  const int tile = blockIdx.x * 4 + wv;         // exact: 50000 tiles
  const int row = lane & 15, g = lane >> 4;
  const int e = tile * 16 + row;
  h16* ar = &sA[wv][row * 32];
  if (g == 0) {
#pragma unroll
    for (int k = 0; k < 5; ++k) ar[k] = (h16)ea[e * 5 + k];
    for (int k = 5; k < 16; ++k) ar[k] = (h16)0.f;
  } else {
    for (int k = 16; k < 32; ++k) ar[k] = (h16)0.f;
  }
  __syncthreads();
  mlp3_tile<32>(sA[wv], sT[wv], sF[wv], w1t, w2t, w3t, b1, b2, b3, gam, bet, lane);
  const float* fr = &sF[wv][row * 32];
  f32x4 hv[4]; read_half_row(fr, g, hv);
#pragma unroll
  for (int q = 0; q < 4; ++q)
    *(h16x4*)(ef + (size_t)e * 32 + 16 * g + 4 * q) = to_h4(hv[q]);
}

__global__ __launch_bounds__(128) void k_mp_edge(
    const h16* __restrict__ nf16, h16* __restrict__ ef,
    const int* __restrict__ ei, const int* __restrict__ dn, MlpW w,
    float* __restrict__ aggr) {
  __shared__ h16 w1t[32 * 96], w2t[32 * 32], w3t[32 * 32];
  __shared__ float b1[32], b2[32], b3[32], gam[32], bet[32];
  __shared__ h16 sA[4][16 * 96], sT[4][16 * 32];
  __shared__ float sF[4][16 * 32];
  const int tid = threadIdx.x, wv = tid >> 5, lane = tid & 31;
  stage_mlp(w, 96, 96, w1t, w2t, w3t, b1, b2, b3, gam, bet, tid, 128);
  const int tile = blockIdx.x * 4 + wv;         // exact: 50000 tiles
  const int row = lane & 15, g = lane >> 4;
  const int e = tile * 16 + row;
  const int dst = ei[N_EDGES + e];
  const int d0 = dn[2 * e], d1 = dn[2 * e + 1];
  const bool dm = (d0 >= 0);
  h16* ar = &sA[wv][row * 96];
  if (g == 0) {                                  // x_i = x[dmask ? d0 : dst]
    int sel = dm ? d0 : dst;
    cp32h(ar, nf16 + (size_t)sel * 32);
  } else {                                       // x_j = x[dmask ? d1 : src]
    int src = ei[e];
    int sel = dm ? d1 : src;
    cp32h(ar + 32, nf16 + (size_t)sel * 32);
  }
  cp16h(ar + 64 + 16 * g, ef + (size_t)e * 32 + 16 * g);   // ef
  __syncthreads();
  mlp3_tile<96>(sA[wv], sT[wv], sF[wv], w1t, w2t, w3t, b1, b2, b3, gam, bet, lane);
  const float* fr = &sF[wv][row * 32];
  f32x4 hv[4]; read_half_row(fr, g, hv);
  const h16* er = ar + 64 + 16 * g;
  float* ag = aggr + (size_t)dst * 32 + 16 * g;
#pragma unroll
  for (int q = 0; q < 4; ++q) {
    h16x4 e4 = *(const h16x4*)(er + 4 * q);
    h16x4 n4;
#pragma unroll
    for (int j = 0; j < 4; ++j) {
      float m = hv[q][j];
      n4[j] = (h16)((float)e4[j] + m);                 // ef += m
      unsafeAtomicAdd(&ag[4 * q + j], m);              // segment_sum(m, dst)
    }
    *(h16x4*)(ef + (size_t)e * 32 + 16 * g + 4 * q) = n4;
  }
}

__global__ __launch_bounds__(128) void k_mp_node(
    float* __restrict__ nf32, h16* __restrict__ nf16,
    float* __restrict__ aggr, MlpW w) {
  __shared__ h16 w1t[32 * 64], w2t[32 * 32], w3t[32 * 32];
  __shared__ float b1[32], b2[32], b3[32], gam[32], bet[32];
  __shared__ h16 sA[4][16 * 64], sT[4][16 * 32];
  __shared__ float sF[4][16 * 32];
  const int tid = threadIdx.x, wv = tid >> 5, lane = tid & 31;
  stage_mlp(w, 64, 64, w1t, w2t, w3t, b1, b2, b3, gam, bet, tid, 128);
  int tile = blockIdx.x * 4 + wv; if (tile >= NODE_TILES) tile = NODE_TILES - 1;
  const int row = lane & 15, g = lane >> 4;
  const int node = tile * 16 + row;
  h16* ar = &sA[wv][row * 64];
  if (g == 0) {
    cp32h(ar, nf16 + (size_t)node * 32);               // x
  } else {
#pragma unroll
    for (int q = 0; q < 8; ++q) {
      f32x4 a4 = *(const f32x4*)(aggr + (size_t)node * 32 + 4 * q);
      *(h16x4*)(ar + 32 + 4 * q) = to_h4(a4);          // aggr -> f16
    }
  }
  __syncthreads();
  mlp3_tile<64>(sA[wv], sT[wv], sF[wv], w1t, w2t, w3t, b1, b2, b3, gam, bet, lane);
  const float* fr = &sF[wv][row * 32];
  f32x4 hv[4]; read_half_row(fr, g, hv);
  const f32x4 z4 = {};
#pragma unroll
  for (int q = 0; q < 4; ++q) {
    int k = 16 * g + 4 * q;
    f32x4 x4 = *(const f32x4*)(nf32 + (size_t)node * 32 + k);
    f32x4 v4 = x4 + hv[q];                             // residual
    *(f32x4*)(nf32 + (size_t)node * 32 + k) = v4;
    *(h16x4*)(nf16 + (size_t)node * 32 + k) = to_h4(v4);
    *(f32x4*)(aggr + (size_t)node * 32 + k) = z4;      // reset for next layer
  }
}

__global__ __launch_bounds__(128) void k_node_out(
    const h16* __restrict__ nf16, MlpW w,
    float* __restrict__ out, float* __restrict__ netI) {
  __shared__ h16 w1t[32 * 32], w2t[32 * 32];
  __shared__ float w3[32], b1[32], b2[32], b3[1];
  __shared__ h16 sA[4][16 * 32], sT[4][16 * 32];
  __shared__ float sF[4][16 * 32];
  const int tid = threadIdx.x, wv = tid >> 5, lane = tid & 31;
  for (int i = tid; i < 1024; i += 128) {
    int n = i >> 5, k = i & 31;
    int c = ((k & 1) << 4) | (k >> 1);
    w1t[i] = (h16)w.W0[k * 32 + n];      // layer-1 K = input space
    w2t[i] = (h16)w.W1[c * 32 + n];      // layer-2 K = permuted hidden
  }
  for (int i = tid; i < 32; i += 128) { w3[i] = w.W2[i]; b1[i] = w.b0[i]; b2[i] = w.b1[i]; }
  if (tid == 0) b3[0] = w.b2[0];
  int tile = blockIdx.x * 4 + wv; if (tile >= NODE_TILES) tile = NODE_TILES - 1;
  const int row = lane & 15, g = lane >> 4, col = lane & 15;
  const int node = tile * 16 + row;
  if (g == 0) cp32h(&sA[wv][row * 32], nf16 + (size_t)node * 32);
  __syncthreads();
  const v8f vz = {};
  v16h a = load_A(sA[wv], 32, 0, lane);
  v8f c0 = wmma16(a, load_B(w1t, 32, 0, 0, lane), vz);
  v8f c1 = wmma16(a, load_B(w1t, 32, 0, 1, lane), vz);
  store_hidden_relu(c0, c1, b1, sT[wv], lane);
  __syncthreads();
  a = load_A(sT[wv], 32, 0, lane);
  c0 = wmma16(a, load_B(w2t, 32, 0, 0, lane), vz);
  c1 = wmma16(a, load_B(w2t, 32, 0, 1, lane), vz);
  {
    float bb0 = b2[col], bb1 = b2[col + 16];
#pragma unroll
    for (int r = 0; r < 8; ++r) {
      int rw = r + 8 * g;
      sF[wv][rw * 32 + col]      = fmaxf(c0[r] + bb0, 0.f);
      sF[wv][rw * 32 + col + 16] = fmaxf(c1[r] + bb1, 0.f);
    }
  }
  __syncthreads();
  if (g == 0) {                                   // final 32 -> 1
    const float* fr = &sF[wv][row * 32];
    float acc = b3[0];
#pragma unroll
    for (int k = 0; k < 32; ++k) acc += fr[k] * w3[k];
    out[node] = acc;
    netI[node] = 0.f;
  }
}

__global__ __launch_bounds__(256) void k_circuit(
    const float* __restrict__ out, const float* __restrict__ ea,
    const int* __restrict__ ei, const int* __restrict__ dn,
    float* __restrict__ netI) {
  const int e = blockIdx.x * 256 + threadIdx.x;
  if (e >= N_EDGES) return;
  const int src = ei[e], dst = ei[N_EDGES + e];
  const int d0 = dn[2 * e], d1 = dn[2 * e + 1];
  const float IL = ea[e * 5 + 0], cond = ea[e * 5 + 1], lI0 = ea[e * 5 + 2];
  const float nid = ea[e * 5 + 3], bd = ea[e * 5 + 4];
  float I = cond * (out[src] - out[dst]) + IL;
  if (d0 >= 0) {
    float Vd = out[d1] - out[d0];
    float sgn = (lI0 > 0.f) ? 1.f : ((lI0 < 0.f) ? -1.f : 0.f);
    float I0 = -sgn * __expf(-fabsf(lI0));
    float corr = I0 * (__expf((Vd - bd) / (nid * VTH)) - 1.f);
    I -= corr;
  }
  unsafeAtomicAdd(&netI[dst], I);
}

__global__ __launch_bounds__(256) void k_node_err(
    const float* __restrict__ netI, const float* __restrict__ y,
    float* __restrict__ err) {
  const int i = blockIdx.x * 256 + threadIdx.x;
  if (i >= N_NODES) return;
  float t = y[i * 3 + 0];
  float v = netI[i];
  err[i] = (t == 1.f) ? 0.f : ((t == 2.f) ? v - y[i * 3 + 2] : v);
}

// ---------------------------------------------------------------------------
// Launch
// ---------------------------------------------------------------------------
extern "C" void kernel_launch(void* const* d_in, const int* in_sizes, int n_in,
                              void* d_out, int out_size, void* d_ws, size_t ws_size,
                              hipStream_t stream) {
  (void)in_sizes; (void)n_in; (void)out_size; (void)ws_size;
  // --- workspace carve (~68 MB) ---
  char* p = (char*)d_ws;
  auto carve = [&](size_t bytes) -> void* {
    void* r = (void*)p; p += (bytes + 255) & ~(size_t)255; return r;
  };
  float* nf32 = (float*)carve((size_t)N_NODES * 32 * 4);
  h16*   nf16 = (h16*)  carve((size_t)N_NODES * 32 * 2);
  float* aggr = (float*)carve((size_t)N_NODES * 32 * 4);
  float* netI = (float*)carve((size_t)N_NODES * 4);
  h16*   ef   = (h16*)  carve((size_t)N_EDGES * 32 * 2);

  // --- inputs: top level in setup_inputs() insertion order; params flattened
  //     as JAX tree leaves (dict keys alphabetical; per-MLP: W0..2,b0..2,be,g)
  const int*   x_type = (const int*)  d_in[0];
  const float* x_feat = (const float*)d_in[1];
  const float* e_attr = (const float*)d_in[2];
  const int*   e_idx  = (const int*)  d_in[3];
  const int*   diode  = (const int*)  d_in[4];
  const float* y      = (const float*)d_in[5];
  auto grab = [&](int base, bool ln) {
    MlpW m;
    m.W0 = (const float*)d_in[base + 0];
    m.W1 = (const float*)d_in[base + 1];
    m.W2 = (const float*)d_in[base + 2];
    m.b0 = (const float*)d_in[base + 3];
    m.b1 = (const float*)d_in[base + 4];
    m.b2 = (const float*)d_in[base + 5];
    m.be = ln ? (const float*)d_in[base + 6] : nullptr;
    m.g  = ln ? (const float*)d_in[base + 7] : nullptr;
    return m;
  };
  MlpW edge_in = grab(6, true);
  const float* embed = (const float*)d_in[14];
  MlpW mpe[4], mpn[4];
  for (int i = 0; i < 4; ++i) { mpe[i] = grab(15 + 16 * i, true); mpn[i] = grab(23 + 16 * i, true); }
  MlpW node_in  = grab(79, true);
  MlpW node_out = grab(87, false);
  float* outv = (float*)d_out;            // [0,50000) = out, [50000,100000) = err

  k_node_encode<<<NODE_BLOCKS, 128, 0, stream>>>(x_type, x_feat, embed, node_in,
                                                 nf32, nf16, aggr);
  k_edge_in<<<EDGE_BLOCKS, 128, 0, stream>>>(e_attr, edge_in, ef);
  for (int l = 0; l < 4; ++l) {
    k_mp_edge<<<EDGE_BLOCKS, 128, 0, stream>>>(nf16, ef, e_idx, diode, mpe[l], aggr);
    k_mp_node<<<NODE_BLOCKS, 128, 0, stream>>>(nf32, nf16, aggr, mpn[l]);
  }
  k_node_out<<<NODE_BLOCKS, 128, 0, stream>>>(nf16, node_out, outv, netI);
  k_circuit<<<(N_EDGES + 255) / 256, 256, 0, stream>>>(outv, e_attr, e_idx, diode, netI);
  k_node_err<<<(N_NODES + 255) / 256, 256, 0, stream>>>(netI, y, outv + N_NODES);
}
